// Model_39676907880912
// MI455X (gfx1250) — compile-verified, matrix operand order checked by reading, not run on hardware
//
#include <hip/hip_runtime.h>
#include <hip/hip_bf16.h>

typedef __attribute__((ext_vector_type(16))) _Float16 v16h;
typedef __attribute__((ext_vector_type(8)))  _Float16 v8h;
typedef __attribute__((ext_vector_type(8)))  float    v8f;
typedef __attribute__((ext_vector_type(4)))  float    v4f;

#define BATCH 4
#define SEQ   4096
#define DIM   64
#define LOG2E 1.4426950408889634f

// Load a 16(row)x32(K) f16 operand in the gfx1250 WMMA A/B register layout.
// rowbase points at the start of this lane's row (row-major, K contiguous).
// lane<16: halves 0..7 = K[ck+0..7],  halves 8..15 = K[ck+16..23]
// lane>=16: halves 0..7 = K[ck+8..15], halves 8..15 = K[ck+24..31]
__device__ __forceinline__ v16h ld_op_f16(const _Float16* rowbase, int ck, int hi) {
    const _Float16* p = rowbase + ck + hi * 8;
    v8h lo = *(const v8h*)(p);
    v8h hp = *(const v8h*)(p + 16);
    return __builtin_shufflevector(lo, hp, 0,1,2,3,4,5,6,7,8,9,10,11,12,13,14,15);
}

// Same layout, but source data is f32 in memory (convert on load).
__device__ __forceinline__ v16h ld_op_f32(const float* rowbase, int ck, int hi) {
    const float* p = rowbase + ck + hi * 8;
    v4f a = *(const v4f*)(p);
    v4f b = *(const v4f*)(p + 4);
    v4f c = *(const v4f*)(p + 16);
    v4f d = *(const v4f*)(p + 20);
    v16h o;
#pragma unroll
    for (int i = 0; i < 4; ++i) {
        o[i]      = (_Float16)a[i];
        o[4 + i]  = (_Float16)b[i];
        o[8 + i]  = (_Float16)c[i];
        o[12 + i] = (_Float16)d[i];
    }
    return o;
}

__device__ __forceinline__ v8f wmma_f16(v16h a, v16h b, v8f c) {
    return __builtin_amdgcn_wmma_f32_16x16x32_f16(false, a, false, b, (short)0, c, false, false);
}

// ---------------------------------------------------------------------------
// Kernel 1: QKV projections. Y = X @ W^T + b (torch Linear convention) via
// v_wmma_f32_16x16x32_f16, output converted to f16.
//   Q is additionally pre-scaled by log2(e) so the attention softmax can use
//   bare v_exp_f32 (2^x) without a per-score multiply.
//   Q,K row-major [B,S,64]; V transposed [B,64,S] for contiguous A-operand
//   loads in the attention kernel.
// One wave = one 16-row tile of one matrix. 3*B*S/16 = 3072 waves.
// ---------------------------------------------------------------------------
__global__ void __launch_bounds__(256) proj_qkv_kernel(
    const float* __restrict__ xq, const float* __restrict__ xk, const float* __restrict__ xv,
    const float* __restrict__ Wq, const float* __restrict__ bq,
    const float* __restrict__ Wk, const float* __restrict__ bk,
    const float* __restrict__ Wv, const float* __restrict__ bv,
    _Float16* __restrict__ Qh, _Float16* __restrict__ Kh, _Float16* __restrict__ VTh)
{
    const int lane = threadIdx.x & 31;
    const int wid  = blockIdx.x * (blockDim.x >> 5) + (threadIdx.x >> 5);
    const int mat  = wid >> 10;          // 0=q, 1=k, 2=v   (1024 tiles each)
    const int tile = wid & 1023;
    const int b    = tile >> 8;          // tile / 256
    const int m0   = (tile & 255) << 4;  // 16-row tile start

    const float* X;  const float* W;  const float* bs;
    if (mat == 0)      { X = xq; W = Wq; bs = bq; }
    else if (mat == 1) { X = xk; W = Wk; bs = bk; }
    else               { X = xv; W = Wv; bs = bv; }
    const float oscale = (mat == 0) ? LOG2E : 1.0f;

    const int r  = lane & 15;
    const int hi = (lane >> 4) & 1;

    // A operand: X rows (m0+r), K = input dim, two 32-chunks
    const float* xrow = X + ((size_t)b * SEQ + m0 + r) * DIM;
    v16h a0 = ld_op_f32(xrow, 0,  hi);
    v16h a1 = ld_op_f32(xrow, 32, hi);

#pragma unroll
    for (int nt = 0; nt < 4; ++nt) {
        const int n = nt * 16 + r;                 // output column for this lane
        const float* wrow = W + (size_t)n * DIM;   // W[n, :] (Linear weight is [out,in])
        v16h w0 = ld_op_f32(wrow, 0,  hi);
        v16h w1 = ld_op_f32(wrow, 32, hi);
        v8f acc = {};
        acc = wmma_f16(a0, w0, acc);
        acc = wmma_f16(a1, w1, acc);
        const float bb = bs[n];

        if (mat < 2) {
            // Row-major f16 store: lane owns column n, rows m0 + j + 8*hi
            _Float16* dst = (mat == 0) ? Qh : Kh;
#pragma unroll
            for (int j = 0; j < 8; ++j) {
                dst[((size_t)b * SEQ + m0 + j + 8 * hi) * DIM + n] =
                    (_Float16)((acc[j] + bb) * oscale);
            }
        } else {
            // V^T store: VT[b][n][s]; rows s = m0 + 8*hi + j are contiguous -> one b128
            v8h pk;
#pragma unroll
            for (int j = 0; j < 8; ++j) pk[j] = (_Float16)(acc[j] + bb);
            *(v8h*)(VTh + ((size_t)b * DIM + n) * SEQ + m0 + 8 * hi) = pk;
        }
    }
}

// ---------------------------------------------------------------------------
// Kernel 2: flash attention, one wave per 16 queries, 64 keys/iteration:
//   T = K_tile(16x64) @ Q^T            -> scores transposed [key, query]
//   online softmax over keys (in-lane + one shfl_xor(16); scores are already
//   in log2(e) units so p = v_exp_f32(s - m) directly)
//   O^T tiles  += V^T_tile(16x32) @ P^T(32x16)  (two K-chunks per d-tile)
// T's f32 C-layout is bit-for-bit the f16 B-operand layout of the second
// WMMA, so P needs only an in-lane f32->f16 convert. 16 WMMAs / 64 keys.
// ---------------------------------------------------------------------------
__global__ void __launch_bounds__(128) attn_kernel(
    const _Float16* __restrict__ Qh, const _Float16* __restrict__ Kh,
    const _Float16* __restrict__ VTh, float* __restrict__ out)
{
    const int lane = threadIdx.x & 31;
    const int wid  = blockIdx.x * (blockDim.x >> 5) + (threadIdx.x >> 5); // 0..1023
    const int b    = wid >> 8;
    const int q0   = (wid & 255) << 4;

    const int r  = lane & 15;
    const int hi = (lane >> 4) & 1;

    const _Float16* Qb  = Qh  + (size_t)b * SEQ * DIM;
    const _Float16* Kb  = Kh  + (size_t)b * SEQ * DIM;
    const _Float16* VTb = VTh + (size_t)b * DIM * SEQ;

    // B operand for the score GEMM: lane holds Q row (q0+r), d in two 32-chunks.
    const _Float16* qrow = Qb + (size_t)(q0 + r) * DIM;
    const v16h qb0 = ld_op_f16(qrow, 0,  hi);
    const v16h qb1 = ld_op_f16(qrow, 32, hi);

    v8f o[4] = {v8f{}, v8f{}, v8f{}, v8f{}};
    float m_run = -__builtin_inff();
    float l_run = 0.0f;

    for (int t0 = 0; t0 < SEQ; t0 += 64) {
        // Prefetch next iteration's K rows for this lane (global_prefetch_b8).
#pragma unroll
        for (int kt = 0; kt < 4; ++kt)
            __builtin_prefetch(Kb + (size_t)(t0 + 64 + kt * 16 + r) * DIM, 0, 3);

        // --- scores (transposed, log2e units): four 16-key tiles ---
        v8f s[4];
#pragma unroll
        for (int kt = 0; kt < 4; ++kt) {
            const _Float16* krow = Kb + (size_t)(t0 + kt * 16 + r) * DIM;
            v16h k0 = ld_op_f16(krow, 0,  hi);
            v16h k1 = ld_op_f16(krow, 32, hi);
            v8f acc = {};
            acc = wmma_f16(k0, qb0, acc);
            acc = wmma_f16(k1, qb1, acc);
            s[kt] = acc;  // s[kt][j] = score(key = t0 + kt*16 + j + 8*hi, query = q0+r)
        }

        // --- online softmax over keys (per query column) ---
        float mloc = -__builtin_inff();
#pragma unroll
        for (int kt = 0; kt < 4; ++kt)
#pragma unroll
            for (int j = 0; j < 8; ++j) mloc = fmaxf(mloc, s[kt][j]);
        mloc = fmaxf(mloc, __shfl_xor(mloc, 16, 32));   // combine hi/lo key halves
        const float m_new = fmaxf(m_run, mloc);
        const float rsc   = __builtin_amdgcn_exp2f(m_run - m_new);

        float ssum = 0.0f;
#pragma unroll
        for (int kt = 0; kt < 4; ++kt)
#pragma unroll
            for (int j = 0; j < 8; ++j) {
                const float p = __builtin_amdgcn_exp2f(s[kt][j] - m_new);
                s[kt][j] = p;
                ssum += p;
            }
        ssum += __shfl_xor(ssum, 16, 32);
        l_run = l_run * rsc + ssum;
        m_run = m_new;

#pragma unroll
        for (int dt = 0; dt < 4; ++dt)
#pragma unroll
            for (int j = 0; j < 8; ++j) o[dt][j] *= rsc;

        // --- P^T as two f16 B operands: pure in-lane convert (layouts coincide) ---
        v16h pB0, pB1;
#pragma unroll
        for (int j = 0; j < 8; ++j) {
            pB0[j]     = (_Float16)s[0][j];
            pB0[8 + j] = (_Float16)s[1][j];
            pB1[j]     = (_Float16)s[2][j];
            pB1[8 + j] = (_Float16)s[3][j];
        }

        // --- O^T accumulate: A = V^T rows (d = dt*16 + r), 2 key-chunks of 32 ---
#pragma unroll
        for (int dt = 0; dt < 4; ++dt) {
            const _Float16* vrow = VTb + (size_t)(dt * 16 + r) * SEQ + t0;
            o[dt] = wmma_f16(ld_op_f16(vrow, 0,  hi), pB0, o[dt]);
            o[dt] = wmma_f16(ld_op_f16(vrow, 32, hi), pB1, o[dt]);
        }
    }

    // --- normalize and store: lane owns query q0+r, d = dt*16 + 8*hi + j ---
    const float inv = 1.0f / l_run;
    float* orow = out + ((size_t)b * SEQ + q0 + r) * DIM + 8 * hi;
#pragma unroll
    for (int dt = 0; dt < 4; ++dt) {
        v4f lo, hp;
#pragma unroll
        for (int j = 0; j < 4; ++j) { lo[j] = o[dt][j] * inv; hp[j] = o[dt][4 + j] * inv; }
        *(v4f*)(orow + dt * 16)     = lo;
        *(v4f*)(orow + dt * 16 + 4) = hp;
    }
}

extern "C" void kernel_launch(void* const* d_in, const int* in_sizes, int n_in,
                              void* d_out, int out_size, void* d_ws, size_t ws_size,
                              hipStream_t stream) {
    (void)in_sizes; (void)n_in; (void)out_size; (void)ws_size;
    const float* query = (const float*)d_in[0];
    const float* key   = (const float*)d_in[1];
    const float* value = (const float*)d_in[2];
    const float* Wq    = (const float*)d_in[3];
    const float* bq    = (const float*)d_in[4];
    const float* Wk    = (const float*)d_in[5];
    const float* bk    = (const float*)d_in[6];
    const float* Wv    = (const float*)d_in[7];
    const float* bv    = (const float*)d_in[8];
    float* out = (float*)d_out;

    const size_t elems = (size_t)BATCH * SEQ * DIM;  // 1,048,576
    _Float16* Qh  = (_Float16*)d_ws;
    _Float16* Kh  = Qh + elems;
    _Float16* VTh = Kh + elems;  // 3 * 2 MB = 6 MB of workspace

    // 3072 waves, 8 waves/block -> 384 blocks
    proj_qkv_kernel<<<384, 256, 0, stream>>>(query, key, value,
                                             Wq, bq, Wk, bk, Wv, bv,
                                             Qh, Kh, VTh);
    // 1024 waves, 4 waves/block -> 256 blocks
    attn_kernel<<<256, 128, 0, stream>>>(Qh, Kh, VTh, out);
}
